// ContextualLoss_3D_34308198761206
// MI455X (gfx1250) — matrix operand order repaired and checked by reading, not compiled, not run on hardware
//
#include <hip/hip_runtime.h>
#include <hip/hip_bf16.h>
#include <math.h>

// CDNA5 / gfx1250, wave32. WMMA bf16 path.
typedef __attribute__((ext_vector_type(16))) __bf16 v16bf;
typedef __attribute__((ext_vector_type(8)))  float  v8f;

#define N_BATCH 8
#define C_DIM   128
#define L_DIM   4096
#define L_TILES 256           // L_DIM / 16
#define CL_EPS  1e-5f
#define WAVES_PER_BLOCK 8
#define TILES_PER_WAVE  (L_TILES / WAVES_PER_BLOCK)   // 32, compile-time uniform
#define TILE_STRIDE     ((size_t)WAVES_PER_BLOCK * 16 * C_DIM)  // halfwords between a wave's tiles

union Frag16 {
    v16bf  v;
    float4 q[2];
};

static __device__ __forceinline__ unsigned short f2bf(float f) {
    unsigned int u = __float_as_uint(f);
    u += 0x7FFFu + ((u >> 16) & 1u);   // round-to-nearest-even
    return (unsigned short)(u >> 16);
}

// ---------------------------------------------------------------------------
// Kernel 1: per-channel mean of y over (batch, spatial):  mu[c]
// ---------------------------------------------------------------------------
__global__ void k_channel_mean(const float* __restrict__ y, float* __restrict__ mu) {
    int c   = blockIdx.x;
    int tid = threadIdx.x;
    float s = 0.f;
    for (int n = 0; n < N_BATCH; ++n) {
        const float* p = y + (size_t)(n * C_DIM + c) * L_DIM;
        for (int l = tid; l < L_DIM; l += 256) s += p[l];
    }
    __shared__ float red[256];
    red[tid] = s;
    __syncthreads();
    for (int off = 128; off > 0; off >>= 1) {
        if (tid < off) red[tid] += red[tid + off];
        __syncthreads();
    }
    if (tid == 0) mu[c] = red[0] / (float)(N_BATCH * L_DIM);
}

// ---------------------------------------------------------------------------
// Kernel 2: center by mu, L2-normalize along C, write transposed bf16 copies
//           xnT/ynT layout: [n][l][c]  (c contiguous -> b128 fragment loads)
// ---------------------------------------------------------------------------
__global__ void k_normalize(const float* __restrict__ x, const float* __restrict__ y,
                            const float* __restrict__ mu,
                            unsigned short* __restrict__ xnT,
                            unsigned short* __restrict__ ynT) {
    __shared__ float mu_s[C_DIM];
    int tid = threadIdx.x;
    if (tid < C_DIM) mu_s[tid] = mu[tid];
    __syncthreads();

    int g = blockIdx.x * 256 + tid;        // flat (n,l), 32768 total
    int n = g >> 12;
    int l = g & (L_DIM - 1);
    const float* xb = x + (size_t)n * C_DIM * L_DIM + l;
    const float* yb = y + (size_t)n * C_DIM * L_DIM + l;

    float sx = 0.f, sy = 0.f;
    for (int c = 0; c < C_DIM; ++c) {
        float m  = mu_s[c];
        float xv = xb[(size_t)c * L_DIM] - m;
        float yv = yb[(size_t)c * L_DIM] - m;
        sx += xv * xv;
        sy += yv * yv;
    }
    float ix = 1.f / fmaxf(sqrtf(sx), 1e-12f);
    float iy = 1.f / fmaxf(sqrtf(sy), 1e-12f);

    unsigned int* xo = (unsigned int*)(xnT + (size_t)g * C_DIM);
    unsigned int* yo = (unsigned int*)(ynT + (size_t)g * C_DIM);
    for (int c = 0; c < C_DIM; c += 2) {
        float m0 = mu_s[c], m1 = mu_s[c + 1];
        float x0 = (xb[(size_t)c       * L_DIM] - m0) * ix;
        float x1 = (xb[(size_t)(c + 1) * L_DIM] - m1) * ix;
        float y0 = (yb[(size_t)c       * L_DIM] - m0) * iy;
        float y1 = (yb[(size_t)(c + 1) * L_DIM] - m1) * iy;
        xo[c >> 1] = (unsigned int)f2bf(x0) | ((unsigned int)f2bf(x1) << 16);
        yo[c >> 1] = (unsigned int)f2bf(y0) | ((unsigned int)f2bf(y1) << 16);
    }
}

// ---------------------------------------------------------------------------
// WMMA tile helper: cos 16x16 tile = A(16 rows l, K=128) x B(K=128, 16 cols m)
// A lane layout: row = lane&15, 16 contiguous K at (lane>>4)*16 per 32-K block
// B lane layout: col = lane&15, 16 contiguous K at (lane>>4)*16 per 32-K block
// ---------------------------------------------------------------------------
static __device__ __forceinline__ v8f cos_tile(const unsigned short* __restrict__ xrow,
                                               const unsigned short* __restrict__ ycol) {
    v8f acc = {0.f, 0.f, 0.f, 0.f, 0.f, 0.f, 0.f, 0.f};
#pragma unroll
    for (int kb = 0; kb < 4; ++kb) {
        Frag16 aF, bF;
        aF.q[0] = *(const float4*)(xrow + kb * 32);
        aF.q[1] = *(const float4*)(xrow + kb * 32 + 8);
        bF.q[0] = *(const float4*)(ycol + kb * 32);
        bF.q[1] = *(const float4*)(ycol + kb * 32 + 8);
        acc = __builtin_amdgcn_wmma_f32_16x16x32_bf16(false, aF.v, false, bF.v,
                                                      (short)0, acc, false, false);
    }
    return acc;
}

// ---------------------------------------------------------------------------
// Kernel 3: rowmax[n,l] = max_m cos[n,l,m]   (dist_min = 1 - rowmax)
// Block = (l-tile, n); 8 waves stripe the 256 m-tiles (uniform 32-iter loop).
// ---------------------------------------------------------------------------
__global__ void __launch_bounds__(256) k_rowmax(const unsigned short* __restrict__ xnT,
                                                const unsigned short* __restrict__ ynT,
                                                float* __restrict__ rowmax) {
    int n = blockIdx.y;
    int l0 = blockIdx.x * 16;
    int tid = threadIdx.x;
    int lane = tid & 31, wave = tid >> 5;
    int col = lane & 15, half = lane >> 4;

    const unsigned short* xrow = xnT + (size_t)(n * L_DIM + l0 + col) * C_DIM + half * 16;
    const unsigned short* ycol =
        ynT + (size_t)(n * L_DIM + wave * 16 + col) * C_DIM + half * 16;

    float mx[8];
#pragma unroll
    for (int i = 0; i < 8; ++i) mx[i] = -INFINITY;

#pragma unroll 2
    for (int it = 0; it < TILES_PER_WAVE; ++it) {
        v8f acc = cos_tile(xrow, ycol);
#pragma unroll
        for (int i = 0; i < 8; ++i) mx[i] = fmaxf(mx[i], acc[i]);
        ycol += TILE_STRIDE;
    }
#pragma unroll
    for (int off = 1; off < 16; off <<= 1) {
#pragma unroll
        for (int i = 0; i < 8; ++i) mx[i] = fmaxf(mx[i], __shfl_xor(mx[i], off, 32));
    }
    __shared__ float red[WAVES_PER_BLOCK][16];
    if (col == 0) {
#pragma unroll
        for (int v = 0; v < 8; ++v) red[wave][half * 8 + v] = mx[v];
    }
    __syncthreads();
    if (tid < 16) {
        float m = red[0][tid];
#pragma unroll
        for (int w = 1; w < WAVES_PER_BLOCK; ++w) m = fmaxf(m, red[w][tid]);
        rowmax[n * L_DIM + l0 + tid] = m;
    }
}

// ---------------------------------------------------------------------------
// Kernel 4: Z[n,l] = sum_m exp(a_l * (cos - rowmax_l)),  a_l = 2/(1-rowmax+eps)
// (shift-invariant softmax denominator; numerator max is exp(0)=1)
// ---------------------------------------------------------------------------
__global__ void __launch_bounds__(256) k_rowsum(const unsigned short* __restrict__ xnT,
                                                const unsigned short* __restrict__ ynT,
                                                const float* __restrict__ rowmax,
                                                float* __restrict__ Z) {
    int n = blockIdx.y;
    int l0 = blockIdx.x * 16;
    int tid = threadIdx.x;
    int lane = tid & 31, wave = tid >> 5;
    int col = lane & 15, half = lane >> 4;

    const unsigned short* xrow = xnT + (size_t)(n * L_DIM + l0 + col) * C_DIM + half * 16;
    const unsigned short* ycol =
        ynT + (size_t)(n * L_DIM + wave * 16 + col) * C_DIM + half * 16;

    const float* rptr = rowmax + n * L_DIM + l0 + half * 8;
    float4 r0 = *(const float4*)(rptr);
    float4 r1 = *(const float4*)(rptr + 4);
    float rmv[8] = {r0.x, r0.y, r0.z, r0.w, r1.x, r1.y, r1.z, r1.w};
    float av[8];
#pragma unroll
    for (int i = 0; i < 8; ++i) av[i] = 2.f / (1.f - rmv[i] + CL_EPS);

    float sm[8];
#pragma unroll
    for (int i = 0; i < 8; ++i) sm[i] = 0.f;

#pragma unroll 2
    for (int it = 0; it < TILES_PER_WAVE; ++it) {
        v8f acc = cos_tile(xrow, ycol);
#pragma unroll
        for (int i = 0; i < 8; ++i) sm[i] += __expf(av[i] * (acc[i] - rmv[i]));
        ycol += TILE_STRIDE;
    }
#pragma unroll
    for (int off = 1; off < 16; off <<= 1) {
#pragma unroll
        for (int i = 0; i < 8; ++i) sm[i] += __shfl_xor(sm[i], off, 32);
    }
    __shared__ float red[WAVES_PER_BLOCK][16];
    if (col == 0) {
#pragma unroll
        for (int v = 0; v < 8; ++v) red[wave][half * 8 + v] = sm[v];
    }
    __syncthreads();
    if (tid < 16) {
        float s = red[0][tid];
#pragma unroll
        for (int w = 1; w < WAVES_PER_BLOCK; ++w) s += red[w][tid];
        Z[n * L_DIM + l0 + tid] = s;
    }
}

// ---------------------------------------------------------------------------
// Kernel 5: colmax[n,m] = max_l exp(a_l*(cos - rowmax_l)) / Z_l
// Block = (m-tile, n); 8 waves stripe the 256 l-tiles (uniform 32-iter loop).
// ---------------------------------------------------------------------------
__global__ void __launch_bounds__(256) k_colmax(const unsigned short* __restrict__ xnT,
                                                const unsigned short* __restrict__ ynT,
                                                const float* __restrict__ rowmax,
                                                const float* __restrict__ Z,
                                                float* __restrict__ colmax) {
    int n = blockIdx.y;
    int m0 = blockIdx.x * 16;
    int tid = threadIdx.x;
    int lane = tid & 31, wave = tid >> 5;
    int col = lane & 15, half = lane >> 4;

    const unsigned short* ycol = ynT + (size_t)(n * L_DIM + m0 + col) * C_DIM + half * 16;
    const unsigned short* xrow =
        xnT + (size_t)(n * L_DIM + wave * 16 + col) * C_DIM + half * 16;
    const float* rptr = rowmax + n * L_DIM + wave * 16 + half * 8;
    const float* zptr = Z + n * L_DIM + wave * 16 + half * 8;

    float cm = 0.f;  // probabilities > 0
#pragma unroll 2
    for (int it = 0; it < TILES_PER_WAVE; ++it) {
        v8f acc = cos_tile(xrow, ycol);

        float4 r0 = *(const float4*)(rptr);
        float4 r1 = *(const float4*)(rptr + 4);
        float4 z0 = *(const float4*)(zptr);
        float4 z1 = *(const float4*)(zptr + 4);
        float rmv[8] = {r0.x, r0.y, r0.z, r0.w, r1.x, r1.y, r1.z, r1.w};
        float zv[8]  = {z0.x, z0.y, z0.z, z0.w, z1.x, z1.y, z1.z, z1.w};
#pragma unroll
        for (int i = 0; i < 8; ++i) {
            float a = 2.f / (1.f - rmv[i] + CL_EPS);
            float p = __expf(a * (acc[i] - rmv[i])) / zv[i];
            cm = fmaxf(cm, p);
        }
        xrow += TILE_STRIDE;
        rptr += WAVES_PER_BLOCK * 16;
        zptr += WAVES_PER_BLOCK * 16;
    }
    cm = fmaxf(cm, __shfl_xor(cm, 16, 32));   // combine the two row-halves
    __shared__ float red[WAVES_PER_BLOCK][16];
    if (lane < 16) red[wave][lane] = cm;
    __syncthreads();
    if (tid < 16) {
        float m = red[0][tid];
#pragma unroll
        for (int w = 1; w < WAVES_PER_BLOCK; ++w) m = fmaxf(m, red[w][tid]);
        colmax[n * L_DIM + m0 + tid] = m;
    }
}

// ---------------------------------------------------------------------------
// Kernel 6: loss = mean_n( -log( mean_m colmax[n,m] + eps ) )
// ---------------------------------------------------------------------------
__global__ void k_loss(const float* __restrict__ colmax, float* __restrict__ out) {
    __shared__ float red[256];
    int tid = threadIdx.x;
    float loss = 0.f;
    for (int n = 0; n < N_BATCH; ++n) {
        float s = 0.f;
        for (int i = tid; i < L_DIM; i += 256) s += colmax[n * L_DIM + i];
        red[tid] = s;
        __syncthreads();
        for (int off = 128; off > 0; off >>= 1) {
            if (tid < off) red[tid] += red[tid + off];
            __syncthreads();
        }
        if (tid == 0) {
            float cx = red[0] / (float)L_DIM;
            loss += -logf(cx + CL_EPS);
        }
        __syncthreads();
    }
    if (tid == 0) out[0] = loss / (float)N_BATCH;
}

// ---------------------------------------------------------------------------
extern "C" void kernel_launch(void* const* d_in, const int* in_sizes, int n_in,
                              void* d_out, int out_size, void* d_ws, size_t ws_size,
                              hipStream_t stream) {
    (void)in_sizes; (void)n_in; (void)out_size; (void)ws_size;
    const float* x = (const float*)d_in[0];
    const float* y = (const float*)d_in[1];
    float* out = (float*)d_out;

    // workspace layout (~16.4 MB total)
    float* mu      = (float*)d_ws;                 // 128
    float* rowmax  = mu + 128;                     // 32768
    float* Zs      = rowmax + N_BATCH * L_DIM;     // 32768
    float* colmax  = Zs + N_BATCH * L_DIM;         // 32768
    unsigned short* xnT = (unsigned short*)(colmax + N_BATCH * L_DIM);  // 8 MB
    unsigned short* ynT = xnT + (size_t)N_BATCH * L_DIM * C_DIM;        // 8 MB

    k_channel_mean<<<C_DIM, 256, 0, stream>>>(y, mu);
    k_normalize<<<(N_BATCH * L_DIM) / 256, 256, 0, stream>>>(x, y, mu, xnT, ynT);

    dim3 g(L_TILES, N_BATCH);
    k_rowmax<<<g, 256, 0, stream>>>(xnT, ynT, rowmax);
    k_rowsum<<<g, 256, 0, stream>>>(xnT, ynT, rowmax, Zs);
    k_colmax<<<g, 256, 0, stream>>>(xnT, ynT, rowmax, Zs, colmax);
    k_loss<<<1, 256, 0, stream>>>(colmax, out);
}
